// BLCRF_35871566856621
// MI455X (gfx1250) — matrix operand order repaired and checked by reading, not compiled, not run on hardware
//
#include <hip/hip_runtime.h>
#include <hip/hip_bf16.h>
#include <math.h>

// Problem constants
#define VSZ 30000
#define DD  256     // embedding dim == 2H (layer-1 input dim)
#define HH  128     // hidden
#define LL  2       // layers
#define KK  32      // tags
#define BB  16      // batch
#define TT  256     // time
#define G4H 512     // 4*H
#define MROWS (TT*BB)   // 4096 rows in time-major activation matrix

typedef __attribute__((ext_vector_type(16))) _Float16 v16h;
typedef __attribute__((ext_vector_type(8)))  float    v8f;

// ---------------------------------------------------------------------------
// WMMA tile loaders, contiguous k order k = 16*(lane/16) + i in A and B (wave32, 16x16x32 f16)
// ---------------------------------------------------------------------------
// A matrix 16x32 (MxK), row-major source with leading dim ld:
//   lanes 0-15:  M=lane, K = 0..15
//   lanes 16-31: M=lane-16, K = 16..31
__device__ __forceinline__ v16h load_a_16x32(const _Float16* __restrict__ p,
                                             int ld, int lane) {
    const int m    = lane & 15;
    const int koff = (lane >> 4) * 16;
    const _Float16* row = p + m * ld;
    v16h a;
#pragma unroll
    for (int i = 0; i < 16; ++i) {
        int k = i + koff;
        a[i] = row[k];
    }
    return a;
}

// B matrix 32x16 (KxN), where the source is stored [N][K] row-major (k contig),
// i.e. B[k][n] = W[n*ld + k].  Layout: lanes 0-15 hold N=lane, K=0..15;
// lanes 16-31 hold N=lane-16, K=16..31.
__device__ __forceinline__ v16h load_b_rowK(const _Float16* __restrict__ w,
                                            int ld, int lane) {
    const int n     = lane & 15;
    const int kbase = (lane >> 4) * 16;
    const _Float16* col = w + n * ld;
    v16h b;
#pragma unroll
    for (int i = 0; i < 16; ++i) b[i] = col[kbase + i];
    return b;
}

// B matrix 32x16 (KxN) where source is stored [K][N] row-major (n contig):
// B[k][n] = W[k*ld + n].
__device__ __forceinline__ v16h load_b_colK(const _Float16* __restrict__ w,
                                            int ld, int lane) {
    const int n     = lane & 15;
    const int kbase = (lane >> 4) * 16;
    v16h b;
#pragma unroll
    for (int i = 0; i < 16; ++i) b[i] = w[(kbase + i) * ld + n];
    return b;
}

__device__ __forceinline__ float sigf(float x) { return 1.0f / (1.0f + __expf(-x)); }

__device__ __forceinline__ float wave_sum32(float v) {
#pragma unroll
    for (int o = 16; o > 0; o >>= 1) v += __shfl_xor(v, o, 32);
    return v;
}
__device__ __forceinline__ float wave_max32(float v) {
#pragma unroll
    for (int o = 16; o > 0; o >>= 1) v = fmaxf(v, __shfl_xor(v, o, 32));
    return v;
}

// ---------------------------------------------------------------------------
// Kernel 1: f32 -> f16 weight conversion (grid-stride)
// ---------------------------------------------------------------------------
__global__ void k_f32_to_f16(const float* __restrict__ src,
                             _Float16* __restrict__ dst, int n) {
    for (int i = blockIdx.x * blockDim.x + threadIdx.x; i < n;
         i += gridDim.x * blockDim.x)
        dst[i] = (_Float16)src[i];
}

// ---------------------------------------------------------------------------
// Kernel 2: embedding gather, produce x[t][b][d] in f16 (time-major)
// ---------------------------------------------------------------------------
__global__ void k_embed(const int* __restrict__ inputs,   // [B][T]
                        const float* __restrict__ emb,    // [V][D]
                        _Float16* __restrict__ x) {       // [T][B][D]
    const int n = TT * BB * DD;
    for (int i = blockIdx.x * blockDim.x + threadIdx.x; i < n;
         i += gridDim.x * blockDim.x) {
        int t = i / (BB * DD);
        int r = i - t * (BB * DD);
        int b = r / DD;
        int d = r - b * DD;
        int tok = inputs[b * TT + t];
        x[i] = (_Float16)emb[(size_t)tok * DD + d];
    }
}

// ---------------------------------------------------------------------------
// Kernel 3: hoisted input projection  Gpre[dir][t*16+b][n] = x @ Wih^T + bias
//   A: x f16 [4096,256]   B: Wih f16 [dir][512][256] (row-major, k contig)
//   one wave per 16x16 C tile, 8 K-chunks of 32
// ---------------------------------------------------------------------------
__global__ void k_input_gemm(const _Float16* __restrict__ x,    // [4096][256]
                             const _Float16* __restrict__ Wih2, // [2][512][256]
                             const float* __restrict__ bih2,    // [2][512]
                             const float* __restrict__ bhh2,    // [2][512]
                             float* __restrict__ Gpre2) {       // [2][4096][512]
    const int mt   = blockIdx.x;          // 0..255
    const int nt   = blockIdx.y;          // 0..31
    const int dir  = blockIdx.z;          // 0..1
    const int lane = threadIdx.x & 31;

    const _Float16* W = Wih2 + (size_t)dir * G4H * DD;
    float* G = Gpre2 + (size_t)dir * MROWS * G4H;

    const int n     = nt * 16 + (lane & 15);
    const int mbase = mt * 16 + ((lane >> 4) * 8);

    v8f acc;
    {
        float bias = bih2[dir * G4H + n] + bhh2[dir * G4H + n];
#pragma unroll
        for (int r = 0; r < 8; ++r) acc[r] = bias;
    }
#pragma unroll
    for (int kc = 0; kc < 8; ++kc) {
        v16h a = load_a_16x32(x + (size_t)mt * 16 * DD + kc * 32, DD, lane);
        v16h b = load_b_rowK(W + (size_t)nt * 16 * DD + kc * 32, DD, lane);
        acc = __builtin_amdgcn_wmma_f32_16x16x32_f16(false, a, false, b,
                                                     (short)0, acc, false, false);
    }
#pragma unroll
    for (int r = 0; r < 8; ++r) G[(size_t)(mbase + r) * G4H + n] = acc[r];
}

// ---------------------------------------------------------------------------
// Kernel 4: sequential LSTM recurrence, one workgroup per direction.
//   per step: g = Gpre[t] + h @ Whh^T  (WMMA, 8 waves x 4 N-tiles x 4 K-chunks)
//             then gates/state update in LDS, h written to next-layer x (f16).
// ---------------------------------------------------------------------------
__global__ __launch_bounds__(256)
void k_lstm_rec(const float* __restrict__ Gpre2,   // [2][4096][512]
                const _Float16* __restrict__ Whh2, // [2][512][128]
                _Float16* __restrict__ xnext) {    // [4096][256] (fwd cols 0..127, bwd 128..255)
    const int dir  = blockIdx.x;
    const int tid  = threadIdx.x;
    const int lane = tid & 31;
    const int wave = tid >> 5;

    const float*    G  = Gpre2 + (size_t)dir * MROWS * G4H;
    const _Float16* Wd = Whh2 + (size_t)dir * G4H * HH;

    __shared__ _Float16 h_s[BB * HH];   // 4 KB
    __shared__ float    c_s[BB * HH];   // 8 KB
    __shared__ float    g_s[BB * G4H];  // 32 KB

    for (int i = tid; i < BB * HH; i += 256) {
        h_s[i] = (_Float16)0.0f;
        c_s[i] = 0.0f;
    }
    __syncthreads();

    const int n_in_tile = lane & 15;
    const int mbase     = (lane >> 4) * 8;

    for (int step = 0; step < TT; ++step) {
        const int t = dir ? (TT - 1 - step) : step;
        const float* Gt = G + (size_t)t * BB * G4H;

        // ---- recurrent GEMM phase: waves cover 32 N-tiles -------------------
#pragma unroll
        for (int q = 0; q < 4; ++q) {
            const int nt = wave * 4 + q;
            const int n  = nt * 16 + n_in_tile;
            v8f acc;
#pragma unroll
            for (int r = 0; r < 8; ++r)
                acc[r] = Gt[(size_t)(mbase + r) * G4H + n];
            // prefetch the weight stream (L2-resident; emits global_prefetch_b8)
            __builtin_prefetch(Wd + (size_t)nt * 16 * HH, 0, 1);
#pragma unroll
            for (int kc = 0; kc < 4; ++kc) {
                v16h a = load_a_16x32(h_s + kc * 32, HH, lane);
                v16h b = load_b_rowK(Wd + (size_t)nt * 16 * HH + kc * 32, HH, lane);
                acc = __builtin_amdgcn_wmma_f32_16x16x32_f16(
                    false, a, false, b, (short)0, acc, false, false);
            }
#pragma unroll
            for (int r = 0; r < 8; ++r)
                g_s[(mbase + r) * G4H + n] = acc[r];
        }
        __syncthreads();

        // ---- gate / state update phase: 2048 (b,j) units over 256 threads ---
#pragma unroll
        for (int q = 0; q < 8; ++q) {
            const int u = tid + q * 256;
            const int b = u >> 7;        // batch row
            const int j = u & 127;       // hidden unit
            const float gi = g_s[b * G4H + j];
            const float gf = g_s[b * G4H + HH + j];
            const float gg = g_s[b * G4H + 2 * HH + j];
            const float go = g_s[b * G4H + 3 * HH + j];
            float c = sigf(gf) * c_s[b * HH + j] + sigf(gi) * tanhf(gg);
            float h = sigf(go) * tanhf(c);
            c_s[b * HH + j] = c;
            h_s[b * HH + j] = (_Float16)h;
            xnext[(size_t)(t * BB + b) * DD + dir * HH + j] = (_Float16)h;
        }
        __syncthreads();
    }
}

// ---------------------------------------------------------------------------
// Kernel 5: emissions = h @ W_out + b_out   ([4096,256] x [256,32])
//   W_out f16 stored [K=256][N=32] row-major -> colK loader
// ---------------------------------------------------------------------------
__global__ void k_emit_gemm(const _Float16* __restrict__ x,     // [4096][256]
                            const _Float16* __restrict__ Wout,  // [256][32]
                            const float* __restrict__ b_out,    // [32]
                            float* __restrict__ em) {           // [4096][32]
    const int mt   = blockIdx.x;   // 0..255
    const int nt   = blockIdx.y;   // 0..1
    const int lane = threadIdx.x & 31;

    const int n     = nt * 16 + (lane & 15);
    const int mbase = mt * 16 + ((lane >> 4) * 8);

    v8f acc;
    {
        float bias = b_out[n];
#pragma unroll
        for (int r = 0; r < 8; ++r) acc[r] = bias;
    }
#pragma unroll
    for (int kc = 0; kc < 8; ++kc) {
        v16h a = load_a_16x32(x + (size_t)mt * 16 * DD + kc * 32, DD, lane);
        v16h b = load_b_colK(Wout + (size_t)kc * 32 * KK + nt * 16, KK, lane);
        acc = __builtin_amdgcn_wmma_f32_16x16x32_f16(false, a, false, b,
                                                     (short)0, acc, false, false);
    }
#pragma unroll
    for (int r = 0; r < 8; ++r) em[(size_t)(mbase + r) * KK + n] = acc[r];
}

// ---------------------------------------------------------------------------
// Kernel 6: CRF gold score + forward algorithm + loss.
//   One block, 16 waves (one per batch row), lane = tag (K=32 == wave32).
// ---------------------------------------------------------------------------
__global__ __launch_bounds__(512)
void k_crf(const float* __restrict__ em,          // [T*16][32], row = t*16+b
           const float* __restrict__ mask,        // [B][T]
           const int*   __restrict__ tags,        // [B][T]
           const float* __restrict__ transitions, // [32][32]
           const float* __restrict__ start_trans, // [32]
           const float* __restrict__ end_trans,   // [32]
           float* __restrict__ out) {
    const int tid  = threadIdx.x;
    const int lane = tid & 31;
    const int b    = tid >> 5;

    __shared__ float tr[KK * KK];
    __shared__ float lossb[BB];
    for (int i = tid; i < KK * KK; i += 512) tr[i] = transitions[i];
    __syncthreads();

    // ---- sequence length + gold score (time strided across lanes) ----------
    float msum = 0.0f, part = 0.0f;
    for (int t = lane; t < TT; t += 32) {
        float mt = mask[b * TT + t];
        msum += mt;
        int tg = tags[b * TT + t];
        part += em[(size_t)(t * BB + b) * KK + tg] * mt;
        if (t >= 1) part += tr[tags[b * TT + t - 1] * KK + tg] * mt;
    }
    msum = wave_sum32(msum);
    part = wave_sum32(part);
    float score = 0.0f;
    if (lane == 0) {
        int len = (int)(msum + 0.5f);
        if (len < 1) len = 1;
        score = part + start_trans[tags[b * TT + 0]]
                     + end_trans[tags[b * TT + len - 1]];
    }

    // ---- forward algorithm: lane k holds alpha[k] ---------------------------
    float alpha = start_trans[lane] + em[(size_t)(0 * BB + b) * KK + lane];
    for (int t = 1; t < TT; ++t) {
        float m = -3.4e38f;
#pragma unroll 8
        for (int j = 0; j < KK; ++j)
            m = fmaxf(m, __shfl(alpha, j, 32) + tr[j * KK + lane]);
        float s = 0.0f;
#pragma unroll 8
        for (int j = 0; j < KK; ++j)
            s += __expf(__shfl(alpha, j, 32) + tr[j * KK + lane] - m);
        float nw = m + __logf(s) + em[(size_t)(t * BB + b) * KK + lane];
        alpha = (mask[b * TT + t] > 0.5f) ? nw : alpha;
    }
    float v  = alpha + end_trans[lane];
    float mv = wave_max32(v);
    float sv = wave_sum32(__expf(v - mv));
    if (lane == 0) lossb[b] = (mv + __logf(sv)) - score;
    __syncthreads();
    if (tid == 0) {
        float s = 0.0f;
#pragma unroll
        for (int i = 0; i < BB; ++i) s += lossb[i];
        out[0] = s / (float)BB;
    }
}

// ---------------------------------------------------------------------------
// Host orchestration
// ---------------------------------------------------------------------------
extern "C" void kernel_launch(void* const* d_in, const int* in_sizes, int n_in,
                              void* d_out, int out_size, void* d_ws, size_t ws_size,
                              hipStream_t stream) {
    (void)in_sizes; (void)n_in; (void)out_size; (void)ws_size;
    const int*   inputs      = (const int*)d_in[0];
    const float* mask        = (const float*)d_in[1];
    const int*   tags        = (const int*)d_in[2];
    const float* emb         = (const float*)d_in[3];
    const float* Wih         = (const float*)d_in[4];   // [2][2][512][256]
    const float* Whh         = (const float*)d_in[5];   // [2][2][512][128]
    const float* bih         = (const float*)d_in[6];   // [2][2][512]
    const float* bhh         = (const float*)d_in[7];   // [2][2][512]
    const float* W_out       = (const float*)d_in[8];   // [256][32]
    const float* b_out       = (const float*)d_in[9];   // [32]
    const float* transitions = (const float*)d_in[10];  // [32][32]
    const float* start_trans = (const float*)d_in[11];  // [32]
    const float* end_trans   = (const float*)d_in[12];  // [32]
    float* out = (float*)d_out;

    // Workspace layout (bytes, 256-aligned)
    char* ws = (char*)d_ws;
    const size_t WIH_N  = (size_t)LL * 2 * G4H * DD;   // 524288
    const size_t WHH_N  = (size_t)LL * 2 * G4H * HH;   // 262144
    const size_t WOUT_N = (size_t)DD * KK;             // 8192
    size_t off = 0;
    _Float16* Wih_h  = (_Float16*)(ws + off); off += WIH_N  * 2;
    _Float16* Whh_h  = (_Float16*)(ws + off); off += WHH_N  * 2;
    _Float16* Wout_h = (_Float16*)(ws + off); off += WOUT_N * 2; off = (off + 255) & ~(size_t)255;
    _Float16* xA     = (_Float16*)(ws + off); off += (size_t)MROWS * DD * 2;
    _Float16* xB     = (_Float16*)(ws + off); off += (size_t)MROWS * DD * 2;
    float*    Gpre   = (float*)(ws + off);    off += (size_t)2 * MROWS * G4H * 4;
    float*    emit   = (float*)(ws + off);    off += (size_t)MROWS * KK * 4;

    // Weight conversions
    k_f32_to_f16<<<dim3((unsigned)((WIH_N + 255) / 256)), 256, 0, stream>>>(Wih, Wih_h, (int)WIH_N);
    k_f32_to_f16<<<dim3((unsigned)((WHH_N + 255) / 256)), 256, 0, stream>>>(Whh, Whh_h, (int)WHH_N);
    k_f32_to_f16<<<dim3((unsigned)((WOUT_N + 255) / 256)), 256, 0, stream>>>(W_out, Wout_h, (int)WOUT_N);

    // Embedding gather -> xA (f16, time-major)
    k_embed<<<dim3(2048), 512, 0, stream>>>(inputs, emb, xA);

    // BiLSTM layers
    for (int l = 0; l < LL; ++l) {
        k_input_gemm<<<dim3(MROWS / 16, G4H / 16, 2), 32, 0, stream>>>(
            xA, Wih_h + (size_t)l * 2 * G4H * DD,
            bih + (size_t)l * 2 * G4H, bhh + (size_t)l * 2 * G4H, Gpre);
        k_lstm_rec<<<dim3(2), 256, 0, stream>>>(
            Gpre, Whh_h + (size_t)l * 2 * G4H * HH, xB);
        _Float16* tmp = xA; xA = xB; xB = tmp;
    }

    // Emissions
    k_emit_gemm<<<dim3(MROWS / 16, KK / 16), 32, 0, stream>>>(xA, Wout_h, b_out, emit);

    // CRF loss
    k_crf<<<dim3(1), 512, 0, stream>>>(emit, mask, tags, transitions,
                                       start_trans, end_trans, out);
}
